// Decoder_48378511622555
// MI455X (gfx1250) — matrix operand order repaired and checked by reading, not compiled, not run on hardware
//
#include <hip/hip_runtime.h>
#include <hip/hip_bf16.h>

#define NC_ 25000
#define NF_ 100000
#define E0_ 400000
#define E1_ 1600000
#define H_  128

typedef __attribute__((ext_vector_type(16))) __bf16 v16bf;
typedef __attribute__((ext_vector_type(4)))  __bf16 bf4;
typedef __attribute__((ext_vector_type(8)))  float  v8f;

__device__ __forceinline__ float elu1(float v) {
    return v > 0.f ? v : (__expf(v) - 1.f);
}

// Convert+transpose a 128x128 fp32 weight (row-major [k][n]) into bf16 [n][k].
__global__ void w_to_bf16_t(const float* __restrict__ W, __bf16* __restrict__ Wt) {
    const int t = blockIdx.x * blockDim.x + threadIdx.x;   // 16384 threads
    const int k = t >> 7, n = t & 127;
    Wt[n * 128 + k] = (__bf16)W[t];                        // t = k*128 + n
}

// ---------------------------------------------------------------------------
// Generic 128-wide GEMM: Y[M,128] = act( X[M,128] @ W[128,128] [+bias] [+Y] )
// - W arrives pre-transposed bf16 ([n][k]); staged to padded LDS via
//   global_load_async_to_lds_b128 (ASYNCcnt / s_wait_asynccnt).
// - X tile converted fp32->bf16 in-register, stored to padded LDS.
// - v_wmma_f32_16x16x32_bf16, fp32 accumulation.
// flags: 1 = add bias, 2 = apply ELU, 4 = accumulate into existing Y
// ---------------------------------------------------------------------------
#define LDSP 136   // padded row pitch (elements) -> conflict-free ds_load_b128
__global__ __launch_bounds__(256)
void gemm128_wmma(const float* __restrict__ X, const __bf16* __restrict__ Wt,
                  const float* __restrict__ bias, float* __restrict__ Y,
                  int M, int flags) {
    __shared__ __bf16 sX[64][LDSP];    // ~17 KB
    __shared__ __bf16 sW[128][LDSP];   // ~34 KB, sW[n][k]
    const int tid = threadIdx.x;
    const int m0  = blockIdx.x * 64;

    // ---- async-DMA stage Wt (128x128 bf16, 2048 x 16B chunks) into padded LDS
    {
        const unsigned lds_base = (unsigned)(uintptr_t)&sW[0][0];
        const unsigned long long gbase = (unsigned long long)(uintptr_t)Wt;
        #pragma unroll
        for (int it = 0; it < 8; ++it) {
            const int c   = tid + it * 256;      // chunk id, 16 chunks/row
            const int row = c >> 4, off = c & 15;
            const unsigned ldst = lds_base + (unsigned)(row * LDSP + off * 8) * 2u;
            const unsigned goff = (unsigned)(row * 128 + off * 8) * 2u;
            asm volatile("global_load_async_to_lds_b128 %0, %1, %2 offset:0"
                         :: "v"(ldst), "v"(goff), "s"(gbase) : "memory");
        }
    }
    // ---- stage X tile (64x128 fp32 -> bf16), rows clamped at the edge
    for (int i = tid; i < 64 * 32; i += 256) {
        const int r = i >> 5, c4 = i & 31;
        int gr = m0 + r; if (gr >= M) gr = M - 1;
        const float4 v = ((const float4*)(X + (size_t)gr * 128))[c4];
        bf4 o = { (__bf16)v.x, (__bf16)v.y, (__bf16)v.z, (__bf16)v.w };
        *(bf4*)&sX[r][c4 * 4] = o;
    }
    asm volatile("s_wait_asynccnt 0x0" ::: "memory");
    __syncthreads();

    const int wave  = tid >> 5;          // 0..7 -> 16-col block
    const int lane  = tid & 31;
    const int n0    = wave * 16;
    const int mlane = lane & 15;
    const int half  = lane >> 4;

    // B fragments (32x16 bf16): lane holds col n = n0+mlane,
    // element i -> K = kbase + half*16 + i  (16 contiguous bf16 = 2x b128)
    v16bf bfrag[4];
    #pragma unroll
    for (int ks = 0; ks < 4; ++ks) {
        const __bf16* p = &sW[n0 + mlane][ks * 32 + half * 16];
        #pragma unroll
        for (int i = 0; i < 16; ++i) bfrag[ks][i] = p[i];
    }

    #pragma unroll
    for (int rb = 0; rb < 4; ++rb) {
        const int rbase = rb * 16;
        v8f c = {};
        #pragma unroll
        for (int ks = 0; ks < 4; ++ks) {
            // A fragment (16x32 bf16): row m = mlane,
            // element i -> K = kbase + half*16 + i
            const __bf16* pr = &sX[rbase + mlane][ks * 32 + half * 16];
            v16bf a;
            #pragma unroll
            for (int i = 0; i < 8; ++i) a[i] = pr[i];
            #pragma unroll
            for (int i = 0; i < 8; ++i) a[8 + i] = pr[8 + i];
            c = __builtin_amdgcn_wmma_f32_16x16x32_bf16(
                    false, a, false, bfrag[ks], (short)0, c, false, false);
        }
        // C/D layout: VGPR r -> M = rbase + r + 8*half, N = lane%16
        #pragma unroll
        for (int r = 0; r < 8; ++r) {
            const int gm = m0 + rbase + r + half * 8;
            if (gm < M) {
                const int idx = gm * 128 + n0 + mlane;
                float v = c[r];
                if (flags & 4) v += Y[idx];
                if (flags & 1) v += bias[n0 + mlane];
                if (flags & 2) v = elu1(v);
                Y[idx] = v;
            }
        }
    }
}

// ---------------------------------------------------------------------------
// Edge-attr MLP: out = elu( elu(a*W0 + b0) @ W1 + b1 ),  1 -> 128 -> 1
// ---------------------------------------------------------------------------
__global__ void edge_mlp(const float* __restrict__ ea, const float* __restrict__ W0,
                         const float* __restrict__ b0, const float* __restrict__ W1,
                         const float* __restrict__ b1, float* __restrict__ out, int E) {
    const int e = blockIdx.x * blockDim.x + threadIdx.x;
    if (e >= E) return;
    const float a = ea[e];
    float acc = 0.f;
    #pragma unroll 8
    for (int j = 0; j < 128; ++j)
        acc += elu1(a * W0[j] + b0[j]) * W1[j];
    out[e] = elu1(acc + b1[0]);
}

__global__ void fill1(float* p, int n) {
    const int t = blockIdx.x * blockDim.x + threadIdx.x;
    if (t < n) p[t] = 1.f;
}
__global__ void deg_scatter(const int* __restrict__ ei, const float* __restrict__ w,
                            float* __restrict__ deg, int E) {
    const int e = blockIdx.x * blockDim.x + threadIdx.x;
    if (e >= E) return;
    atomicAdd(&deg[ei[E + e]], w[e]);
}
__global__ void deg_to_dinv(float* p, int n) {
    const int t = blockIdx.x * blockDim.x + threadIdx.x;
    if (t < n) { const float d = p[t]; p[t] = d > 0.f ? rsqrtf(d) : 0.f; }
}
__global__ void self_init(const float* __restrict__ h, const float* __restrict__ dinv,
                          float* __restrict__ acc, int rows) {
    const int t = blockIdx.x * blockDim.x + threadIdx.x;
    if (t >= rows * 128) return;
    const float d = dinv[t >> 7];
    acc[t] = h[t] * d * d;
}
// acc[col] += h[row] * dinv[row]*w*dinv[col]   (one wave per edge, grid-stride)
__global__ void gcn_scatter(const float* __restrict__ h, const int* __restrict__ ei,
                            const float* __restrict__ ew, const float* __restrict__ dinv,
                            float* __restrict__ acc, int E) {
    const int lane = threadIdx.x & 31;
    const int nwav = (gridDim.x * blockDim.x) >> 5;
    for (int e = (blockIdx.x * blockDim.x + threadIdx.x) >> 5; e < E; e += nwav) {
        const int r = ei[e], c = ei[E + e];
        const float norm = dinv[r] * ew[e] * dinv[c];
        const float* hr = h + (size_t)r * 128;
        float* ac = acc + (size_t)c * 128;
        #pragma unroll
        for (int j = 0; j < 4; ++j)
            atomicAdd(&ac[lane + 32 * j], hr[lane + 32 * j] * norm);
    }
}
__global__ void bias_elu(const float* __restrict__ acc, const float* __restrict__ b,
                         float* __restrict__ out, int rows) {
    const int t = blockIdx.x * blockDim.x + threadIdx.x;
    if (t >= rows * 128) return;
    out[t] = elu1(acc[t] + b[t & 127]);
}
__global__ void c2f_l1(const float* __restrict__ pos0, const float* __restrict__ pos1,
                       const int* __restrict__ cl, const float* __restrict__ W0,
                       const float* __restrict__ b0, float* __restrict__ u, int n) {
    const int f = blockIdx.x, j = threadIdx.x;
    if (f >= n) return;
    const int c = cl[f];
    const float d0 = pos1[c * 2 + 0] - pos0[f * 2 + 0];
    const float d1 = pos1[c * 2 + 1] - pos0[f * 2 + 1];
    u[(size_t)f * 128 + j] = elu1(d0 * W0[j] + d1 * W0[128 + j] + b0[j]);
}
__global__ void gather_rows(const float* __restrict__ src, const int* __restrict__ idx,
                            float* __restrict__ dst, int rows) {
    const int t = blockIdx.x * blockDim.x + threadIdx.x;
    if (t >= rows * 32) return;
    const int f = t >> 5, c = t & 31;
    ((float4*)dst)[(size_t)f * 32 + c] = ((const float4*)src)[(size_t)idx[f] * 32 + c];
}
__global__ void resid_ln(const float* __restrict__ a, const float* __restrict__ bb,
                         const float* __restrict__ g, const float* __restrict__ beta,
                         float* __restrict__ out, int rows) {
    const int row = blockIdx.x * 8 + (threadIdx.x >> 5);
    const int lane = threadIdx.x & 31;
    if (row >= rows) return;
    const float* pa = a + (size_t)row * 128;
    const float* pb = bb + (size_t)row * 128;
    float v[4], s = 0.f;
    #pragma unroll
    for (int j = 0; j < 4; ++j) { v[j] = pa[lane + 32 * j] + pb[lane + 32 * j]; s += v[j]; }
    #pragma unroll
    for (int o = 16; o > 0; o >>= 1) s += __shfl_xor(s, o, 32);
    const float mu = s * (1.f / 128.f);
    float var = 0.f;
    #pragma unroll
    for (int j = 0; j < 4; ++j) { const float d = v[j] - mu; var += d * d; }
    #pragma unroll
    for (int o = 16; o > 0; o >>= 1) var += __shfl_xor(var, o, 32);
    const float rs = rsqrtf(var * (1.f / 128.f) + 1e-5f);
    float* po = out + (size_t)row * 128;
    #pragma unroll
    for (int j = 0; j < 4; ++j)
        po[lane + 32 * j] = (v[j] - mu) * rs * g[lane + 32 * j] + beta[lane + 32 * j];
}
__global__ void rowdot_elu(const float* __restrict__ X, const float* __restrict__ w,
                           const float* __restrict__ b, float* __restrict__ out, int rows) {
    const int row = blockIdx.x * 8 + (threadIdx.x >> 5);
    const int lane = threadIdx.x & 31;
    if (row >= rows) return;
    float s = 0.f;
    #pragma unroll
    for (int j = 0; j < 4; ++j)
        s += X[(size_t)row * 128 + lane + 32 * j] * w[lane + 32 * j];
    #pragma unroll
    for (int o = 16; o > 0; o >>= 1) s += __shfl_xor(s, o, 32);
    if (lane == 0) out[row] = elu1(s + b[0]);
}
__global__ void conv1d3(const float* __restrict__ v, const float* __restrict__ w,
                        const float* __restrict__ b, float* __restrict__ out, int n) {
    const int i = blockIdx.x * blockDim.x + threadIdx.x;
    if (i >= n) return;
    const float l = i > 0 ? v[i - 1] : 0.f;
    const float r = i < n - 1 ? v[i + 1] : 0.f;
    out[i] = w[0] * l + w[1] * v[i] + w[2] * r + b[0];
}
__global__ void int_to_float(const int* __restrict__ s, float* __restrict__ d, int n) {
    const int t = blockIdx.x * blockDim.x + threadIdx.x;
    if (t < n) d[t] = (float)s[t];
}
__global__ void copyf(const float* __restrict__ s, float* __restrict__ d, int n) {
    const int t = blockIdx.x * blockDim.x + threadIdx.x;
    if (t < n) d[t] = s[t];
}

extern "C" void kernel_launch(void* const* d_in, const int* in_sizes, int n_in,
                              void* d_out, int out_size, void* d_ws, size_t ws_size,
                              hipStream_t stream) {
    (void)in_sizes; (void)n_in; (void)out_size; (void)ws_size;
    const float* x      = (const float*)d_in[0];
    const int*   ei0    = (const int*)  d_in[1];
    const float* eat0   = (const float*)d_in[2];
    const int*   ei1    = (const int*)  d_in[3];
    const float* eat1   = (const float*)d_in[4];
    const float* pos0   = (const float*)d_in[6];
    const float* pos1   = (const float*)d_in[7];
    const int*   clus   = (const int*)  d_in[9];
    // params flattened in alphabetical dict-key order (jax pytree):
    const float* c2fW0 = (const float*)d_in[10]; const float* c2fb0 = (const float*)d_in[11];
    const float* c2fW1 = (const float*)d_in[12]; const float* c2fb1 = (const float*)d_in[13];
    const float* convw = (const float*)d_in[14]; const float* convb = (const float*)d_in[15];
    const float* e0W0  = (const float*)d_in[16]; const float* e0b0  = (const float*)d_in[17];
    const float* e0W1  = (const float*)d_in[18]; const float* e0b1  = (const float*)d_in[19];
    const float* e1W0  = (const float*)d_in[20]; const float* e1b0  = (const float*)d_in[21];
    const float* e1W1  = (const float*)d_in[22]; const float* e1b1  = (const float*)d_in[23];
    const float* lng   = (const float*)d_in[24]; const float* lnb   = (const float*)d_in[25];
    const float* m0W0  = (const float*)d_in[26]; const float* m0b0  = (const float*)d_in[27];
    const float* m0W1  = (const float*)d_in[28]; const float* m0b1  = (const float*)d_in[29];
    const float* m1W0  = (const float*)d_in[30]; const float* m1b0  = (const float*)d_in[31];
    const float* m1W1  = (const float*)d_in[32]; const float* m1b1  = (const float*)d_in[33];
    const float* ndW0  = (const float*)d_in[34]; const float* ndb0  = (const float*)d_in[35];
    const float* ndW1  = (const float*)d_in[36]; const float* ndb1  = (const float*)d_in[37];
    const float* upW0  = (const float*)d_in[38]; const float* upb0  = (const float*)d_in[39];
    const float* upW1  = (const float*)d_in[40]; const float* upb1  = (const float*)d_in[41];
    float* out = (float*)d_out;

    // workspace layout (floats)
    float* ws = (float*)d_ws; size_t off = 0;
    float* ea0   = ws + off; off += E0_;
    float* ea1v  = ws + off; off += E1_;
    float* dinvC = ws + off; off += NC_;
    float* dinvF = ws + off; off += NF_;
    float* scal1 = ws + off; off += NF_;
    float* C1 = ws + off; off += (size_t)NC_ * 128;
    float* C2 = ws + off; off += (size_t)NC_ * 128;
    float* C3 = ws + off; off += (size_t)NC_ * 128;
    float* F1 = ws + off; off += (size_t)NF_ * 128;
    float* F2 = ws + off; off += (size_t)NF_ * 128;
    float* F3 = ws + off; off += (size_t)NF_ * 128;
    __bf16* Wtb = (__bf16*)(ws + off); off += 128 * 128 / 2;  // 16384 bf16 scratch

    const int TB = 256;
    #define GR(n) dim3(((n) + TB - 1) / TB)
    const dim3 blk(TB);
    const dim3 gC((NC_ + 63) / 64), gF((NF_ + 63) / 64);
    const dim3 gW(64);                    // 16384 threads for weight convert
    const dim3 gsc(4096);                 // grid-stride scatter
    const int BIAS = 1, ELU = 2, ACC = 4;

    // gemm helper: convert/transpose weight to bf16 scratch, then WMMA gemm
    auto GEMM = [&](const float* Xp, const float* Wp, const float* bp,
                    float* Yp, int M, dim3 grid, int flags) {
        w_to_bf16_t<<<gW, blk, 0, stream>>>(Wp, Wtb);
        gemm128_wmma<<<grid, blk, 0, stream>>>(Xp, Wtb, bp, Yp, M, flags);
    };

    // edge-weight MLPs
    edge_mlp<<<GR(E0_), blk, 0, stream>>>(eat0, e0W0, e0b0, e0W1, e0b1, ea0, E0_);
    edge_mlp<<<GR(E1_), blk, 0, stream>>>(eat1, e1W0, e1b0, e1W1, e1b1, ea1v, E1_);
    // symmetric-norm degrees (self-loop weight 1 folded into init)
    fill1<<<GR(NC_), blk, 0, stream>>>(dinvC, NC_);
    deg_scatter<<<GR(E0_), blk, 0, stream>>>(ei0, ea0, dinvC, E0_);
    deg_to_dinv<<<GR(NC_), blk, 0, stream>>>(dinvC, NC_);
    fill1<<<GR(NF_), blk, 0, stream>>>(dinvF, NF_);
    deg_scatter<<<GR(E1_), blk, 0, stream>>>(ei1, ea1v, dinvF, E1_);
    deg_to_dinv<<<GR(NF_), blk, 0, stream>>>(dinvF, NF_);

    // ---- coarse GCN layer 1 ----
    GEMM(x, m0W0, nullptr, C1, NC_, gC, 0);
    self_init<<<GR(NC_ * 128), blk, 0, stream>>>(C1, dinvC, C2, NC_);
    gcn_scatter<<<gsc, blk, 0, stream>>>(C1, ei0, ea0, dinvC, C2, E0_);
    bias_elu<<<GR(NC_ * 128), blk, 0, stream>>>(C2, m0b0, C2, NC_);
    // ---- coarse GCN layer 2 ----
    GEMM(C2, m0W1, nullptr, C1, NC_, gC, 0);
    self_init<<<GR(NC_ * 128), blk, 0, stream>>>(C1, dinvC, C3, NC_);
    gcn_scatter<<<gsc, blk, 0, stream>>>(C1, ei0, ea0, dinvC, C3, E0_);
    bias_elu<<<GR(NC_ * 128), blk, 0, stream>>>(C3, m0b1, C2, NC_);   // x_coarse = C2

    // ---- coarse -> fine ----
    c2f_l1<<<dim3(NF_), dim3(128), 0, stream>>>(pos0, pos1, clus, c2fW0, c2fb0, F1, NF_);
    GEMM(F1, c2fW1, c2fb1, F2, NF_, gF, BIAS | ELU);                  // ea_c2f = F2
    gather_rows<<<GR(NF_ * 32), blk, 0, stream>>>(C2, clus, F1, NF_); // xg = F1
    GEMM(F2, upW0, nullptr, F3, NF_, gF, 0);
    GEMM(F1, upW0 + 128 * 128, upb0, F3, NF_, gF, ACC | BIAS | ELU);
    GEMM(F3, upW1, upb1, F1, NF_, gF, BIAS | ELU);                    // t = F1
    resid_ln<<<dim3((NF_ + 7) / 8), blk, 0, stream>>>(F2, F1, lng, lnb, F3, NF_); // x_fine = F3

    // ---- fine GCN layer 1 ----
    GEMM(F3, m1W0, nullptr, F1, NF_, gF, 0);
    self_init<<<GR(NF_ * 128), blk, 0, stream>>>(F1, dinvF, F2, NF_);
    gcn_scatter<<<gsc, blk, 0, stream>>>(F1, ei1, ea1v, dinvF, F2, E1_);
    bias_elu<<<GR(NF_ * 128), blk, 0, stream>>>(F2, m1b0, F2, NF_);
    // ---- fine GCN layer 2 ----
    GEMM(F2, m1W1, nullptr, F1, NF_, gF, 0);
    self_init<<<GR(NF_ * 128), blk, 0, stream>>>(F1, dinvF, F3, NF_);
    gcn_scatter<<<gsc, blk, 0, stream>>>(F1, ei1, ea1v, dinvF, F3, E1_);
    bias_elu<<<GR(NF_ * 128), blk, 0, stream>>>(F3, m1b1, F3, NF_);

    // ---- node MLP + conv1d ----
    GEMM(F3, ndW0, ndb0, F1, NF_, gF, BIAS | ELU);
    rowdot_elu<<<dim3((NF_ + 7) / 8), blk, 0, stream>>>(F1, ndW1, ndb1, scal1, NF_);
    conv1d3<<<GR(NF_), blk, 0, stream>>>(scal1, convw, convb, out, NF_);

    // ---- pass-through outputs ----
    int_to_float<<<GR(2 * E1_), blk, 0, stream>>>(ei1, out + NF_, 2 * E1_);
    copyf<<<GR(E1_), blk, 0, stream>>>(ea1v, out + NF_ + 2 * E1_, E1_);
    #undef GR
}
